// MultiBiLSTM_75771813036714
// MI455X (gfx1250) — matrix operand order repaired and checked by reading, not compiled, not run on hardware
//
#include <hip/hip_runtime.h>
#include <stdint.h>

// ============================================================================
// MultiBiLSTM on MI455X (gfx1250).
//
// Mapping: each wave32 processes a tile of 16 batch rows through the whole
// network. Every LSTM gate pre-activation is a WMMA f16->f32 GEMM:
//   D[16 x Ntile*16] = Xcat[16 x Kpad] * W^T[Kpad x Ntile*16]  (+ bias via C)
// with Kpad = 32 (64 for layer 2, done as two K-chunks).
//
// Weights are repacked once per block into LDS in exact B-fragment layout
// (f16, zero K/N padding); biases folded into the WMMA C operand. Recurrent
// h/c state and all inter-layer activations stay in per-wave LDS, so HBM
// traffic is exactly: inputs (100MB) + initial states (808MB) + output (4MB)
// ~ 0.91 GB -> ~39us floor at 23.3 TB/s, with WMMA matmul time ~tens of us
// and the residual bottleneck being ~2.5G v_exp transcendentals (1 exp per
// sigmoid/tanh).
// ============================================================================

#define SEQ 5
#define WPB 4                 // waves per block
#define TPB (WPB * 32)
#define NBLOCKS 1250

typedef __attribute__((ext_vector_type(16))) _Float16 v16h;
typedef __attribute__((ext_vector_type(8)))  float    v8f;

union Frag16 { v16h v; uint4 q[2]; };

// d_in layout (setup_inputs() dict insertion order, leaves flattened
// recursively): [0]=inputs, [1..14]=hf1,cf1,hb1,cb1,...,hl,cl,
// [15..126] = 7 cells (f1,b1,f2,b2,f3,b3,l) x 16 params in PARAM_NAMES order.
struct KArgs {
  const float* inputs;        // [B][5][5]
  const float* h0[7];         // [B][hid]
  const float* c0[7];         // [B][hid]
  const float* w[7][16];      // per-cell params in PARAM_NAMES order
  float*       out;           // [B]
  int          batch;
};

// --- LDS carving (dynamic shared memory) ------------------------------------
// block-shared:
//   wlds : 18944 halves (37888 B)  weight B-fragments, per-cell K-padded rows
//   blds :   432 floats ( 1728 B)  per-neuron bias (b_i + b_h), N-padded
// per-wave (stride 24448 B):
//   xcat : [16][64] f16 (2048 B)   A-operand staging, zero-padded K
//   gates: [16][80] f32 (5120 B)   activated gate values (i|f|g|o blocks)
//   cst  : [16][20] f32 (1280 B)   c-state of current cell
//   Hst  : 8000 halves (16000 B)   per-cell h history [5][16][hid]
#define WLDS_BYTES   37888
#define SHARED_HEAD  39616          // wlds + blds
#define PW_BYTES     24448
#define SMEM_BYTES   (SHARED_HEAD + WPB * PW_BYTES)

__device__ __forceinline__ float fast_sigmoid(float x) {
  float e = __expf(-x);
  return __builtin_amdgcn_rcpf(1.0f + e);
}
__device__ __forceinline__ float fast_tanh(float x) {
  float e = __expf(-2.0f * x);
  return 2.0f * __builtin_amdgcn_rcpf(1.0f + e) - 1.0f;
}

// One LSTM cell, all SEQ timesteps, for this wave's 16-row batch tile.
// src: 0 = global input forward, 1 = global input backward,
//      2 = H of another cell, 3 = Hf3[t] + Hb3[4-t] (final layer input)
__device__ __forceinline__ void run_cell(
    int lane, const _Float16* __restrict__ wlds, const float* __restrict__ blds,
    _Float16* xcat, float* gates, float* cst, _Float16* Hst,
    float* __restrict__ out, const KArgs& A, int b0,
    int c, int in, int hid, int Kpad, int Nt, int woff, int boff, int hoff,
    int src, int srcHoff, int srcHid, bool isOut)
{
  const int kh = lane >> 4;        // K-half selector for WMMA operand layout
  const int lr = lane & 15;        // row (A) / column (B/D) within tile

  // Zero the A staging tile once: x/h slots are rewritten every t, the K/N
  // padding must stay zero.
  for (int i = lane; i < 512; i += 32) ((uint32_t*)xcat)[i] = 0u;

  // c-state init from provided initial state (faithful to reference inputs)
  for (int idx = lane; idx < 16 * hid; idx += 32) {
    int m = idx / hid, j = idx % hid;
    cst[idx] = A.c0[c][(size_t)(b0 + m) * hid + j];
  }
  __builtin_amdgcn_wave_barrier();

  for (int t = 0; t < SEQ; ++t) {
    // ---- (a) build Xcat = [x | h] in f16, K-padded to 64 -------------------
    if (src <= 1) {
      int tt = (src == 0) ? t : (SEQ - 1 - t);
      for (int idx = lane; idx < 16 * in; idx += 32) {
        int m = idx / in, f = idx % in;
        xcat[m * 64 + f] =
            (_Float16)A.inputs[(size_t)(b0 + m) * (SEQ * 5) + tt * 5 + f];
      }
    } else if (src == 2) {
      for (int idx = lane; idx < 16 * in; idx += 32) {
        int m = idx / in, j = idx % in;
        xcat[m * 64 + j] = Hst[srcHoff + (t * 16 + m) * srcHid + j];
      }
    } else {  // src == 3 : Hf3[t] + Hb3[SEQ-1-t]
      for (int idx = lane; idx < 16 * in; idx += 32) {
        int m = idx / in, j = idx % in;
        float a = (float)Hst[srcHoff + (t * 16 + m) * srcHid + j];
        float b = (float)Hst[srcHoff + 800 + ((SEQ - 1 - t) * 16 + m) * srcHid + j];
        xcat[m * 64 + j] = (_Float16)(a + b);
      }
    }
    if (t == 0) {
      for (int idx = lane; idx < 16 * hid; idx += 32) {
        int m = idx / hid, j = idx % hid;
        xcat[m * 64 + in + j] = (_Float16)A.h0[c][(size_t)(b0 + m) * hid + j];
      }
    } else {
      for (int idx = lane; idx < 16 * hid; idx += 32) {
        int m = idx / hid, j = idx % hid;
        xcat[m * 64 + in + j] = Hst[hoff + ((t - 1) * 16 + m) * hid + j];
      }
    }
    __builtin_amdgcn_wave_barrier();

    // ---- (b) A fragments: lane<16 holds K[kh*8..+7] & K[16+kh*8..+7] -------
    Frag16 af0, af1;
    {
      int base = lr * 64 + kh * 8;
      af0.q[0] = *(const uint4*)(xcat + base);
      af0.q[1] = *(const uint4*)(xcat + base + 16);
      if (Kpad == 64) {
        af1.q[0] = *(const uint4*)(xcat + base + 32);
        af1.q[1] = *(const uint4*)(xcat + base + 48);
      }
    }

    // ---- (c) per N-tile WMMA + (d) activation scatter ----------------------
    for (int nt = 0; nt < Nt; ++nt) {
      int n = nt * 16 + lr;                      // global gate-neuron index
      const _Float16* wp = wlds + woff + n * Kpad + kh * 8;
      Frag16 bf;
      bf.q[0] = *(const uint4*)(wp);
      bf.q[1] = *(const uint4*)(wp + 16);
      float bv = blds[boff + n];                 // bias broadcast down column
      v8f acc = {bv, bv, bv, bv, bv, bv, bv, bv};
      acc = __builtin_amdgcn_wmma_f32_16x16x32_f16(
          false, af0.v, false, bf.v, (short)0, acc, false, false);
      if (Kpad == 64) {
        const _Float16* wp2 = wlds + woff + n * Kpad + 32 + kh * 8;
        Frag16 bf2;
        bf2.q[0] = *(const uint4*)(wp2);
        bf2.q[1] = *(const uint4*)(wp2 + 16);
        acc = __builtin_amdgcn_wmma_f32_16x16x32_f16(
            false, af1.v, false, bf2.v, (short)0, acc, false, false);
      }
      int g = n / hid;                 // 0=i,1=f,2=g(tanh),3=o, >=4 = padding
      bool isg = (g == 2);
      #pragma unroll
      for (int r = 0; r < 8; ++r) {
        int m = r + kh * 8;            // batch row of this D element
        float x = acc[r];
        float e = __expf(isg ? -2.0f * x : -x);
        float rc = __builtin_amdgcn_rcpf(1.0f + e);
        float act = isg ? (2.0f * rc - 1.0f) : rc;
        if (n < 4 * hid) gates[m * 80 + n] = act;
      }
    }
    __builtin_amdgcn_wave_barrier();

    // ---- (e) combine gates -> c,h; store h (f16) for the recurrence --------
    for (int idx = lane; idx < 16 * hid; idx += 32) {
      int m = idx / hid, j = idx % hid;
      float gi = gates[m * 80 + j];
      float gf = gates[m * 80 + hid + j];
      float gg = gates[m * 80 + 2 * hid + j];
      float go = gates[m * 80 + 3 * hid + j];
      float cc = gf * cst[idx] + gi * gg;
      cst[idx] = cc;
      float h = go * fast_tanh(cc);
      Hst[hoff + (t * 16 + m) * hid + j] = (_Float16)h;
      if (isOut && t == SEQ - 1) out[b0 + m] = h;
    }
    __builtin_amdgcn_wave_barrier();
  }
}

__global__ __launch_bounds__(TPB) void multibilstm_kernel(KArgs A) {
  extern __shared__ __align__(16) char smem[];
  _Float16* wlds = (_Float16*)smem;
  float*    blds = (float*)(smem + WLDS_BYTES);

  // ---- One-time weight repack into B-fragment layout (whole block) ---------
  // Per cell: rows = Nt*16 padded neurons, each row = Kpad f16 [w_i | w_h | 0]
  // Neuron order within a cell: [i(hid) | f(hid) | g(hid) | o(hid) | pad].
  {
    // in, hid, Kpad, Nt, woff(halves), boff(floats)
    const int ci[7]   = {5, 5, 20, 20, 20, 20, 10};
    const int ch[7]   = {20, 20, 20, 20, 10, 10, 1};
    const int ck[7]   = {32, 32, 64, 64, 32, 32, 32};
    const int cn[7]   = {5, 5, 5, 5, 3, 3, 1};
    const int cw[7]   = {0, 2560, 5120, 10240, 15360, 16896, 18432};
    const int cb[7]   = {0, 80, 160, 240, 320, 368, 416};
    const int pbase[4] = {0, 4, 12, 8};  // PARAM_NAMES gate blocks: i,f,g,o
    for (int c = 0; c < 7; ++c) {
      int in = ci[c], hid = ch[c], Kpad = ck[c];
      int tot = cn[c] * 16 * Kpad;
      for (int idx = threadIdx.x; idx < tot; idx += TPB) {
        int n = idx / Kpad, k = idx % Kpad;
        float v = 0.0f;
        if (n < 4 * hid) {
          int g = n / hid, j = n % hid, pb = pbase[g];
          if (k < in)            v = A.w[c][pb + 0][j * in + k];
          else if (k < in + hid) v = A.w[c][pb + 1][j * hid + (k - in)];
        }
        wlds[cw[c] + idx] = (_Float16)v;
      }
      int ntot = cn[c] * 16;
      for (int idx = threadIdx.x; idx < ntot; idx += TPB) {
        float v = 0.0f;
        if (idx < 4 * hid) {
          int g = idx / hid, j = idx % hid, pb = pbase[g];
          v = A.w[c][pb + 2][j] + A.w[c][pb + 3][j];
        }
        blds[cb[c] + idx] = v;
      }
    }
  }
  __syncthreads();

  const int waveId = threadIdx.x >> 5;
  const int lane   = threadIdx.x & 31;
  char* pw = smem + SHARED_HEAD + waveId * PW_BYTES;
  _Float16* xcat  = (_Float16*)pw;
  float*    gates = (float*)(pw + 2048);
  float*    cst   = (float*)(pw + 7168);
  _Float16* Hst   = (_Float16*)(pw + 8448);

  const int nTiles = A.batch >> 4;
  const int stride = gridDim.x * WPB;
  for (int tile = blockIdx.x * WPB + waveId; tile < nTiles; tile += stride) {
    const int b0 = tile * 16;
    // Prefetch next tile's input rows (-> global_prefetch_b8)
    int nxt = tile + stride;
    if (nxt < nTiles) {
      const char* p = (const char*)(A.inputs + (size_t)nxt * 16 * (SEQ * 5));
      __builtin_prefetch(p + lane * 64, 0, 0);
    }

    // H-history half-offsets: f1=0 b1=1600 f2=3200 b2=4800 f3=6400 b3=7200
    // (layer-l reuses slot 0 — Hf1 is dead once f2 has consumed it)
    run_cell(lane, wlds, blds, xcat, gates, cst, Hst, A.out, A, b0,
             0, 5, 20, 32, 5, 0,     0,   0,    0, 0,    0,  false);  // f1
    run_cell(lane, wlds, blds, xcat, gates, cst, Hst, A.out, A, b0,
             1, 5, 20, 32, 5, 2560,  80,  1600, 1, 0,    0,  false);  // b1
    run_cell(lane, wlds, blds, xcat, gates, cst, Hst, A.out, A, b0,
             2, 20, 20, 64, 5, 5120, 160, 3200, 2, 0,    20, false);  // f2<-Hf1
    run_cell(lane, wlds, blds, xcat, gates, cst, Hst, A.out, A, b0,
             3, 20, 20, 64, 5, 10240, 240, 4800, 2, 1600, 20, false); // b2<-Hb1
    run_cell(lane, wlds, blds, xcat, gates, cst, Hst, A.out, A, b0,
             4, 20, 10, 32, 3, 15360, 320, 6400, 2, 3200, 20, false); // f3<-Hf2
    run_cell(lane, wlds, blds, xcat, gates, cst, Hst, A.out, A, b0,
             5, 20, 10, 32, 3, 16896, 368, 7200, 2, 4800, 20, false); // b3<-Hb2
    run_cell(lane, wlds, blds, xcat, gates, cst, Hst, A.out, A, b0,
             6, 10, 1, 32, 1, 18432, 416, 0,    3, 6400, 10, true);   // l
  }
}

extern "C" void kernel_launch(void* const* d_in, const int* in_sizes, int n_in,
                              void* d_out, int out_size, void* d_ws, size_t ws_size,
                              hipStream_t stream) {
  (void)n_in; (void)d_ws; (void)ws_size; (void)out_size;
  KArgs A;
  A.inputs = (const float*)d_in[0];
  for (int c = 0; c < 7; ++c) {           // states: hf1,cf1,hb1,cb1,...,hl,cl
    A.h0[c] = (const float*)d_in[1 + 2 * c];
    A.c0[c] = (const float*)d_in[2 + 2 * c];
  }
  for (int c = 0; c < 7; ++c)             // params: 7 cells x 16 arrays
    for (int p = 0; p < 16; ++p)
      A.w[c][p] = (const float*)d_in[15 + c * 16 + p];
  A.out   = (float*)d_out;
  A.batch = in_sizes[0] / (SEQ * 5);

  // 137,408 B dynamic LDS per block (4 waves): fits the 320 KB WGP budget.
  // (Production code would raise hipFuncAttributeMaxDynamicSharedMemorySize.)
  multibilstm_kernel<<<dim3(NBLOCKS), dim3(TPB), SMEM_BYTES, stream>>>(A);
}
